// FlashAttention_78108275245191
// MI455X (gfx1250) — matrix operand order repaired and checked by reading, not compiled, not run on hardware
//
#include <hip/hip_runtime.h>
#include <cmath>

typedef __attribute__((ext_vector_type(16))) _Float16 v16h;
typedef __attribute__((ext_vector_type(8)))  _Float16 v8h;
typedef __attribute__((ext_vector_type(8)))  float    v8f;
typedef __attribute__((ext_vector_type(4)))  float    v4f;

union AFrag {
    v16h v;
    v8h  h[2];
};

// Problem constants (match reference setup_inputs)
constexpr int NB  = 2;      // batch
constexpr int NH  = 16;     // heads
constexpr int SEQ = 4096;   // sequence length
constexpr int HD  = 64;     // head dim
constexpr int QT  = 128;    // query rows per workgroup
constexpr int KT  = 128;    // key rows per LDS tile

__device__ __forceinline__ v8f wmma_f16(const v16h& a, const v16h& b, const v8f& c) {
    return __builtin_amdgcn_wmma_f32_16x16x32_f16(
        /*neg_a=*/false, a, /*neg_b=*/false, b,
        /*c_mod=*/(short)0, c, /*reuse_a=*/false, /*reuse_b=*/false);
}

// Raw v_exp_f32: no libm range fixup. Args here are always <= 0; anything
// below the denormal range should flush to 0, which is what we want.
__device__ __forceinline__ float fast_exp2(float x) {
    return __builtin_amdgcn_exp2f(x);
}

// Raw v_rcp_f32: denominator is a positive well-scaled softmax sum; the
// ~1 ulp error of v_rcp is irrelevant vs the f16 matmul quantization.
__device__ __forceinline__ float fast_rcp(float x) {
    return __builtin_amdgcn_rcpf(x);
}

__launch_bounds__(256, 1)
__global__ void fa_fwd_kernel(const float* __restrict__ Qg,
                              const float* __restrict__ Kg,
                              const float* __restrict__ Vg,
                              float* __restrict__ Og) {
    // LDS: Q tile (dead after fragment build) aliases the P staging buffer.
    //   sQP as Q:  [QT][HD] f16            (16 KB)
    //   sQP as P:  per-wave [16][64] f16   (8*2 KB = 16 KB, same storage)
    __shared__ _Float16 sQP[QT * HD];   // 16 KB
    __shared__ _Float16 sK[KT][HD];     // 16 KB
    __shared__ _Float16 sVt[HD][KT];    // 16 KB, [d][key]

    const int tid  = threadIdx.x;
    const int lane = tid & 31;
    const int wave = tid >> 5;
    const int hi   = lane >> 4;   // which half of the wave (0/1)
    const int ln   = lane & 15;   // lane within half

    const int qb = blockIdx.x;                 // query tile index, 0..SEQ/QT-1
    const int bh = blockIdx.y;                 // fused batch*head
    const size_t base = (size_t)bh * SEQ * HD;

    // fold softmax scale AND log2(e) into Q so softmax runs in base-2 domain
    const float qscale = 1.4426950408889634f / sqrtf((float)HD);

    // ---------------- load + scale + convert Q tile ----------------
    {
        const float* gq = Qg + base + (size_t)qb * QT * HD;
        #pragma unroll
        for (int i = 0; i < 8; ++i) {
            int idx4 = tid + i * 256;          // float4 index within tile
            int row  = idx4 >> 4;              // HD/4 = 16 float4 per row
            int c4   = (idx4 & 15) << 2;
            v4f q4 = *(const v4f*)(gq + row * HD + c4);
            _Float16* d = &sQP[row * HD + c4];
            d[0] = (_Float16)(q4.x * qscale);
            d[1] = (_Float16)(q4.y * qscale);
            d[2] = (_Float16)(q4.z * qscale);
            d[3] = (_Float16)(q4.w * qscale);
        }
    }
    __syncthreads();

    // ---------------- Q A-fragments, kept in registers ----------------
    // A 16x32 f16 layout: lane holds row m = lane%16; elements 0..7 -> K = 8*hi..,
    // elements 8..15 -> K = 8*hi+16..  (chunk c adds 32c; here K == d).
    AFrag aQ0, aQ1;
    {
        const _Float16* qrowp = &sQP[(wave * 16 + ln) * HD];
        const int ks0 = hi * 8;
        aQ0.h[0] = *(const v8h*)(qrowp + ks0);
        aQ0.h[1] = *(const v8h*)(qrowp + ks0 + 16);
        aQ1.h[0] = *(const v8h*)(qrowp + ks0 + 32);
        aQ1.h[1] = *(const v8h*)(qrowp + ks0 + 48);
    }

    // P staging region for this wave (aliases sQP; this wave's own Q rows).
    _Float16* sPw = &sQP[wave * 16 * 64];   // [16 rows][64 keys]

    // ones B-fragment: L += P x ones gives row sums on the matrix pipe
    v16h onesB;
    #pragma unroll
    for (int j = 0; j < 16; ++j) onesB[j] = (_Float16)1.0f;

    // online-softmax state (C layout rows: row = r + 8*hi; stats replicated
    // across each 16-lane group)
    float m_r[8];
    v8f   acc[4];
    v8f   accL;   // running softmax denominator via ones-matmul
    #pragma unroll
    for (int r = 0; r < 8; ++r) m_r[r] = -INFINITY;
    {
        v8f z = {0.f,0.f,0.f,0.f,0.f,0.f,0.f,0.f};
        #pragma unroll
        for (int t = 0; t < 4; ++t) acc[t] = z;
        accL = z;
    }

    const int qrow_base = qb * QT + wave * 16 + hi * 8;  // + r gives this lane's rows

    // ---------------- one 128-key tile; diag is compile-time after inlining ----
    auto tile_body = [&](int kb, bool diag) __attribute__((always_inline)) {
        __syncthreads();  // all waves done with previous K/V tile

        const float* gk = Kg + base + (size_t)kb * KT * HD;
        const float* gv = Vg + base + (size_t)kb * KT * HD;
        #pragma unroll
        for (int i = 0; i < 8; ++i) {
            int idx4 = tid + i * 256;
            int row  = idx4 >> 4;
            int c4   = (idx4 & 15) << 2;
            v4f k4 = *(const v4f*)(gk + row * HD + c4);
            _Float16* dk = &sK[row][c4];
            dk[0] = (_Float16)k4.x; dk[1] = (_Float16)k4.y;
            dk[2] = (_Float16)k4.z; dk[3] = (_Float16)k4.w;
            v4f vv = *(const v4f*)(gv + row * HD + c4);
            sVt[c4 + 0][row] = (_Float16)vv.x;
            sVt[c4 + 1][row] = (_Float16)vv.y;
            sVt[c4 + 2][row] = (_Float16)vv.z;
            sVt[c4 + 3][row] = (_Float16)vv.w;
        }
        // prefetch next tile into cache while this one is consumed
        if (!diag) {
            const float* nk = gk + (size_t)KT * HD;
            const float* nv = gv + (size_t)KT * HD;
            __builtin_prefetch(nk + tid * 32, 0, 1);   // 128 B/thread covers 32 KB tile
            __builtin_prefetch(nv + tid * 32, 0, 1);
        }
        __syncthreads();

        #pragma unroll
        for (int g = 0; g < 2; ++g) {        // 64 keys per softmax pass
            const int k0 = g * 64;

            // ---- scores: four 16x16 tiles over keys k0 .. k0+63 ----
            // B 32x16 f16 layout: lane holds column n = lane%16 (a key);
            // elements 0..7 -> K(dim) = 8*hi.., elements 8..15 -> K = 8*hi+16..
            v8f s[4];
            #pragma unroll
            for (int t2 = 0; t2 < 4; ++t2) {
                const int key = k0 + t2 * 16 + ln;
                AFrag bk;
                bk.h[0] = *(const v8h*)&sK[key][hi * 8];
                bk.h[1] = *(const v8h*)&sK[key][hi * 8 + 16];
                v8f z = {0.f,0.f,0.f,0.f,0.f,0.f,0.f,0.f};
                s[t2] = wmma_f16(aQ0.v, bk.v, z);
                bk.h[0] = *(const v8h*)&sK[key][32 + hi * 8];
                bk.h[1] = *(const v8h*)&sK[key][32 + hi * 8 + 16];
                s[t2] = wmma_f16(aQ1.v, bk.v, s[t2]);
            }

            // ---- causal mask: only instantiated in the peeled diagonal call ----
            if (diag) {
                #pragma unroll
                for (int t2 = 0; t2 < 4; ++t2) {
                    const int key = kb * KT + k0 + t2 * 16 + ln;
                    #pragma unroll
                    for (int r = 0; r < 8; ++r) {
                        if (key > qrow_base + r) s[t2][r] = -INFINITY;
                    }
                }
            }

            // ---- online softmax over 64 keys (base-2 domain) ----
            #pragma unroll
            for (int r = 0; r < 8; ++r) {
                float mx = fmaxf(fmaxf(s[0][r], s[1][r]), fmaxf(s[2][r], s[3][r]));
                mx = fmaxf(mx, __shfl_xor(mx, 1));
                mx = fmaxf(mx, __shfl_xor(mx, 2));
                mx = fmaxf(mx, __shfl_xor(mx, 4));
                mx = fmaxf(mx, __shfl_xor(mx, 8));
                float mnew  = fmaxf(m_r[r], mx);
                float alpha = fast_exp2(m_r[r] - mnew);
                m_r[r] = mnew;
                accL[r] *= alpha;
                #pragma unroll
                for (int t = 0; t < 4; ++t) acc[t][r] *= alpha;
                // exp + stage P into LDS (C layout lane=key-col -> A layout lane=q-row)
                _Float16* prow = sPw + (hi * 8 + r) * 64;
                #pragma unroll
                for (int t2 = 0; t2 < 4; ++t2) {
                    float e = fast_exp2(s[t2][r] - mnew);
                    prow[t2 * 16 + ln] = (_Float16)e;
                }
            }

            // ---- P A-fragments (same-wave DS ops are in-order) ----
            const _Float16* prd = sPw + ln * 64;
            AFrag pA0, pA1;
            pA0.h[0] = *(const v8h*)(prd + hi * 8);
            pA0.h[1] = *(const v8h*)(prd + hi * 8 + 16);
            pA1.h[0] = *(const v8h*)(prd + 32 + hi * 8);
            pA1.h[1] = *(const v8h*)(prd + 32 + hi * 8 + 16);

            // denominator on the matrix pipe: accL += P x ones
            accL = wmma_f16(pA0.v, onesB, accL);
            accL = wmma_f16(pA1.v, onesB, accL);

            // ---- PV: acc[t] += P(16x64) x V(64x16) per 16-wide d tile ----
            #pragma unroll
            for (int t = 0; t < 4; ++t) {
                const _Float16* vrow = &sVt[t * 16 + ln][k0];
                AFrag vB;
                vB.h[0] = *(const v8h*)(vrow + hi * 8);
                vB.h[1] = *(const v8h*)(vrow + hi * 8 + 16);
                acc[t] = wmma_f16(pA0.v, vB.v, acc[t]);
                vB.h[0] = *(const v8h*)(vrow + 32 + hi * 8);
                vB.h[1] = *(const v8h*)(vrow + 32 + hi * 8 + 16);
                acc[t] = wmma_f16(pA1.v, vB.v, acc[t]);
            }
        }
    };

    // steady-state tiles: no mask code at all
    for (int kb = 0; kb < qb; ++kb) tile_body(kb, false);
    // peeled diagonal tile: mask code instantiated only here
    tile_body(qb, true);

    // ---------------- epilogue: normalize and store fp32 output ----------------
    float* go = Og + base + (size_t)(qb * QT + wave * 16) * HD;
    #pragma unroll
    for (int r = 0; r < 8; ++r) {
        float inv = fast_rcp(accL[r]);  // row sum, replicated across 16-lane group
        int row = hi * 8 + r;
        #pragma unroll
        for (int t = 0; t < 4; ++t) {
            go[row * HD + t * 16 + ln] = acc[t][r] * inv;
        }
    }
}

extern "C" void kernel_launch(void* const* d_in, const int* in_sizes, int n_in,
                              void* d_out, int out_size, void* d_ws, size_t ws_size,
                              hipStream_t stream) {
    const float* Q = (const float*)d_in[0];
    const float* K = (const float*)d_in[1];
    const float* V = (const float*)d_in[2];
    float* O = (float*)d_out;

    dim3 grid(SEQ / QT, NB * NH);   // (32, 32) workgroups
    dim3 block(256);                // 8 waves of 32
    fa_fwd_kernel<<<grid, block, 0, stream>>>(Q, K, V, O);
}
